// MultiHeadAttention_70317204570640
// MI455X (gfx1250) — matrix operand order repaired and checked by reading, not compiled, not run on hardware
//
#include <hip/hip_runtime.h>
#include <hip/hip_bf16.h>

#define TSEQ 2048
#define CDIM 1024
#define NB   4
#define NH   16
#define HDD  64
#define BT   (NB * TSEQ)   // 8192

typedef __attribute__((ext_vector_type(16))) __bf16 bf16x16;
typedef __attribute__((ext_vector_type(8)))  float  f32x8;

union Frag {
    bf16x16 v;
    uint4   q[2];
};

struct __align__(8) bf16x4s { __bf16 x, y, z, w; };

__device__ __forceinline__ f32x8 wmma_bf16(bf16x16 a, bf16x16 b, f32x8 c) {
    return __builtin_amdgcn_wmma_f32_16x16x32_bf16(
        false, a, false, b, (short)0, c, false, false);
}

__device__ __forceinline__ float redmax16(float v) {
    v = fmaxf(v, __shfl_xor(v, 1, 32));
    v = fmaxf(v, __shfl_xor(v, 2, 32));
    v = fmaxf(v, __shfl_xor(v, 4, 32));
    v = fmaxf(v, __shfl_xor(v, 8, 32));
    return v;
}
__device__ __forceinline__ float redsum16(float v) {
    v += __shfl_xor(v, 1, 32);
    v += __shfl_xor(v, 2, 32);
    v += __shfl_xor(v, 4, 32);
    v += __shfl_xor(v, 8, 32);
    return v;
}

// ---------------------------------------------------------------------------
// One-time f32 -> bf16 conversion (x and the four weight matrices).
// ---------------------------------------------------------------------------
__global__ __launch_bounds__(256)
void cvt_f32_bf16(const float* __restrict__ in, __bf16* __restrict__ out, int n4) {
    int i = blockIdx.x * 256 + threadIdx.x;
    if (i < n4) {
        float4 f = ((const float4*)in)[i];
        bf16x4s v = { (__bf16)f.x, (__bf16)f.y, (__bf16)f.z, (__bf16)f.w };
        ((bf16x4s*)out)[i] = v;
    }
}

// ---------------------------------------------------------------------------
// Tensor Data Mover: DMA one 128x32 bf16 tile (row stride CDIM elems) into
// LDS with 16B padding every 64B -> 80B LDS row stride (LDSTR=40 bf16).
// D# per CDNA5 ISA.  Tracked by TENSORcnt.
// This toolchain exposes the 6-arg builtin: (g0:u32x4, g1:i32x8, g2:i32x4,
// g3:i32x4, i32x8, cpol) — the trailing i32x8 is zero for 2-D tensors.
// ---------------------------------------------------------------------------
#define LDSTR 40   // bf16 elems per LDS row (64B data + 16B TDM pad)

#if __has_builtin(__builtin_amdgcn_tensor_load_to_lds) && \
    __has_builtin(__builtin_amdgcn_s_wait_tensorcnt)
#define HAVE_TDM 1
typedef __attribute__((ext_vector_type(4))) unsigned int u32x4;
typedef __attribute__((ext_vector_type(8))) int          i32x8;
typedef __attribute__((ext_vector_type(4))) int          i32x4;

__device__ __forceinline__ void tdm_load_tile(const __bf16* gtile, unsigned ldsOff,
                                              int tensorRows) {
    unsigned long long ga = (unsigned long long)(uintptr_t)gtile;
    u32x4 g0;
    g0.x = 1u;                                             // count=1 (user D#)
    g0.y = ldsOff;                                         // LDS byte address
    g0.z = (unsigned)ga;                                   // global_addr[31:0]
    g0.w = (unsigned)((ga >> 32) & 0x01ffffffu) | (2u << 30); // addr[56:32]|type=2
    i32x8 g1;
    g1[0] = (1 << 16)        // data_size = 2 bytes
          | (1 << 20)        // pad_enable
          | (3 << 22)        // pad_interval: every 16 dwords (64B)
          | (3 << 25);       // pad_amount: 4 dwords (16B)
    g1[1] = (int)((CDIM & 0xffff) << 16);                  // tensor_dim0 lo16
    g1[2] = (int)((CDIM >> 16) | ((tensorRows & 0xffff) << 16)); // dim0 hi | dim1 lo
    g1[3] = (int)(((unsigned)tensorRows >> 16) | (32u << 16));   // dim1 hi | tile_dim0=32
    g1[4] = 128;                                           // tile_dim1 = 128
    g1[5] = CDIM;                                          // tensor_dim0_stride
    g1[6] = 0;
    g1[7] = 0;
    i32x4 gz4 = { 0, 0, 0, 0 };
    i32x8 gz8 = { 0, 0, 0, 0, 0, 0, 0, 0 };
    __builtin_amdgcn_tensor_load_to_lds(g0, g1, gz4, gz4, gz8, 0);
}
#else
#define HAVE_TDM 0
#endif

// ---------------------------------------------------------------------------
// NT GEMM:  Out[m,n] = scale * sum_k A[m,k] * W[n,k]   (A,W bf16 row-major)
// 128x128 block tile, 8 waves (4x2), v_wmma_f32_16x16x32_bf16, TDM-fed
// double-buffered LDS.
// mode 0: Q bf16 [B,H,T,HD]   mode 1: K bf16 [B,H,T,HD]
// mode 2: Vt bf16 [B,H,HD,T]  mode 3: f32 [M,N] (final output)
// ---------------------------------------------------------------------------
__device__ __forceinline__ void compute_tile(const __bf16* As, const __bf16* Bs,
                                             int waveM, int waveN, int l16, int half,
                                             f32x8 (&acc)[2][4]) {
    Frag a[2], b[4];
#pragma unroll
    for (int ti = 0; ti < 2; ++ti) {
        // A-frag: lane=M row; elem j -> k = half*16 + j (32B contiguous)
        const __bf16* p = As + (waveM * 32 + ti * 16 + l16) * LDSTR + half * 16;
        a[ti].q[0] = *(const uint4*)p;
        a[ti].q[1] = *(const uint4*)(p + 8);
    }
#pragma unroll
    for (int tj = 0; tj < 4; ++tj) {
        // B-frag: lane=N col; elem j -> k = half*16 + j (32B contiguous)
        const __bf16* p = Bs + (waveN * 64 + tj * 16 + l16) * LDSTR + half * 16;
        b[tj].q[0] = *(const uint4*)p;
        b[tj].q[1] = *(const uint4*)(p + 8);
    }
#pragma unroll
    for (int ti = 0; ti < 2; ++ti)
#pragma unroll
        for (int tj = 0; tj < 4; ++tj)
            acc[ti][tj] = wmma_bf16(a[ti].v, b[tj].v, acc[ti][tj]);
}

__global__ __launch_bounds__(256)
void gemm_nt_bf16wmma(const __bf16* __restrict__ A,   // [M,1024] bf16
                      const __bf16* __restrict__ W,   // [1024,1024] bf16 (N x K)
                      void* __restrict__ Out,
                      float scale, int mode, int mRows)
{
    __shared__ __bf16 AsBuf[2][128 * LDSTR];
    __shared__ __bf16 BsBuf[2][128 * LDSTR];

    const int tid   = threadIdx.x;
    const int lane  = tid & 31;
    const int wave  = tid >> 5;
    const int waveM = wave & 3;    // 0..3 -> 32-row slice
    const int waveN = wave >> 2;   // 0..1 -> 64-col slice
    const int l16   = lane & 15;
    const int half  = lane >> 4;

    const int mBlock = blockIdx.x * 128;
    const int nBlock = blockIdx.y * 128;

    f32x8 acc[2][4];
#pragma unroll
    for (int i = 0; i < 2; ++i)
#pragma unroll
        for (int j = 0; j < 4; ++j)
#pragma unroll
            for (int e = 0; e < 8; ++e) acc[i][j][e] = 0.0f;

#if HAVE_TDM
    int cur = 0;
    if (tid < 32) {   // one wave posts the TDM descriptors (EXEC-independent op)
        tdm_load_tile(A + (size_t)mBlock * CDIM, (unsigned)(uintptr_t)&AsBuf[0][0], mRows);
        tdm_load_tile(W + (size_t)nBlock * CDIM, (unsigned)(uintptr_t)&BsBuf[0][0], CDIM);
    }
    for (int k0 = 0; k0 < CDIM; k0 += 32) {
        if (tid < 32) {
            if (k0 + 32 < CDIM) {   // prefetch next tile pair into the other buffer
                tdm_load_tile(A + (size_t)mBlock * CDIM + k0 + 32,
                              (unsigned)(uintptr_t)&AsBuf[cur ^ 1][0], mRows);
                tdm_load_tile(W + (size_t)nBlock * CDIM + k0 + 32,
                              (unsigned)(uintptr_t)&BsBuf[cur ^ 1][0], CDIM);
                __builtin_amdgcn_s_wait_tensorcnt(2);  // current pair done, next in flight
            } else {
                __builtin_amdgcn_s_wait_tensorcnt(0);
            }
        }
        __syncthreads();
        compute_tile(AsBuf[cur], BsBuf[cur], waveM, waveN, l16, half, acc);
        __syncthreads();
        cur ^= 1;
    }
#else
    for (int k0 = 0; k0 < CDIM; k0 += 32) {
#pragma unroll
        for (int i = 0; i < 2; ++i) {
            int c   = tid + i * 256;   // 0..511 chunks of 16B
            int row = c >> 2;
            int cg  = c & 3;
            uint4 va = *(const uint4*)(A + (size_t)(mBlock + row) * CDIM + k0 + cg * 8);
            uint4 vb = *(const uint4*)(W + (size_t)(nBlock + row) * CDIM + k0 + cg * 8);
            *(uint4*)(AsBuf[0] + row * LDSTR + cg * 8) = va;
            *(uint4*)(BsBuf[0] + row * LDSTR + cg * 8) = vb;
        }
        __syncthreads();
        compute_tile(AsBuf[0], BsBuf[0], waveM, waveN, l16, half, acc);
        __syncthreads();
    }
#endif

    // C-layout: reg r, half -> M = half*8 + r ; N = l16
#pragma unroll
    for (int ti = 0; ti < 2; ++ti) {
#pragma unroll
        for (int tj = 0; tj < 4; ++tj) {
#pragma unroll
            for (int r = 0; r < 8; ++r) {
                int m = mBlock + waveM * 32 + ti * 16 + half * 8 + r;
                int n = nBlock + waveN * 64 + tj * 16 + l16;
                float val = acc[ti][tj][r] * scale;
                if (mode == 3) {
                    ((float*)Out)[(size_t)m * CDIM + n] = val;
                } else {
                    int bb = m >> 11;          // m / TSEQ
                    int t  = m & (TSEQ - 1);
                    int h  = n >> 6;           // n / HDD
                    int d  = n & (HDD - 1);
                    __bf16* O = (__bf16*)Out;
                    if (mode == 2) {   // Vt: [B,H,HD,T]
                        O[((size_t)(bb * NH + h) * HDD + d) * TSEQ + t] = (__bf16)val;
                    } else {           // Q / K: [B,H,T,HD]
                        O[((size_t)(bb * NH + h) * TSEQ + t) * HDD + d] = (__bf16)val;
                    }
                }
            }
        }
    }
}

// ---------------------------------------------------------------------------
// Flash attention (causal, online softmax). Grid (T/128, B*H), 8 waves,
// 16 query rows per wave; 16 WMMA per 64-key block. Writes bf16 [B,T,C].
// ---------------------------------------------------------------------------
__global__ __launch_bounds__(256)
void flash_attn_bf16(const __bf16* __restrict__ Q,
                     const __bf16* __restrict__ Kh,
                     const __bf16* __restrict__ Vt,
                     __bf16* __restrict__ Oattn)
{
    __shared__ __bf16 Ps[8][16][64];   // per-wave private P staging (16 KB)

    const int tid  = threadIdx.x;
    const int lane = tid & 31;
    const int wave = tid >> 5;
    const int l16  = lane & 15;
    const int half = lane >> 4;

    const int bh = blockIdx.y;
    const int b  = bh >> 4;
    const int h  = bh & 15;
    const int qBase = blockIdx.x * 128 + wave * 16;

    const size_t headQK = (size_t)bh * TSEQ * HDD;
    const size_t headV  = (size_t)bh * HDD * TSEQ;

    Frag aq[2];
#pragma unroll
    for (int c = 0; c < 2; ++c) {
        const __bf16* p = Q + headQK + (size_t)(qBase + l16) * HDD + c * 32 + half * 16;
        aq[c].q[0] = *(const uint4*)p;
        aq[c].q[1] = *(const uint4*)(p + 8);
    }

    f32x8 o[4];
#pragma unroll
    for (int tn = 0; tn < 4; ++tn)
#pragma unroll
        for (int e = 0; e < 8; ++e) o[tn][e] = 0.0f;

    float mrun[8], lrun[8];
#pragma unroll
    for (int r = 0; r < 8; ++r) { mrun[r] = -1e30f; lrun[r] = 0.0f; }

    const int nkb = (qBase + 15) / 64 + 1;   // causal
    for (int kb = 0; kb < nkb; ++kb) {
        const int kBase = kb * 64;

        f32x8 s[4];
#pragma unroll
        for (int tj = 0; tj < 4; ++tj)
#pragma unroll
            for (int e = 0; e < 8; ++e) s[tj][e] = 0.0f;

#pragma unroll
        for (int tj = 0; tj < 4; ++tj) {
#pragma unroll
            for (int c = 0; c < 2; ++c) {
                Frag bk;
                const __bf16* p = Kh + headQK + (size_t)(kBase + tj * 16 + l16) * HDD
                                  + c * 32 + half * 16;
                bk.q[0] = *(const uint4*)p;
                bk.q[1] = *(const uint4*)(p + 8);
                s[tj] = wmma_bf16(aq[c].v, bk.v, s[tj]);
            }
        }

        if (kBase + 63 > qBase) {
#pragma unroll
            for (int tj = 0; tj < 4; ++tj)
#pragma unroll
                for (int r = 0; r < 8; ++r) {
                    int kIdx = kBase + tj * 16 + l16;
                    int qIdx = qBase + half * 8 + r;
                    if (kIdx > qIdx) s[tj][r] = -1e30f;
                }
        }

        float alpha[8];
#pragma unroll
        for (int r = 0; r < 8; ++r) {
            float v = fmaxf(fmaxf(s[0][r], s[1][r]), fmaxf(s[2][r], s[3][r]));
            v = redmax16(v);
            float mnew = fmaxf(mrun[r], v);
            alpha[r] = __expf(mrun[r] - mnew);
            mrun[r] = mnew;
        }
#pragma unroll
        for (int tj = 0; tj < 4; ++tj)
#pragma unroll
            for (int r = 0; r < 8; ++r)
                s[tj][r] = __expf(s[tj][r] - mrun[r]);
#pragma unroll
        for (int r = 0; r < 8; ++r) {
            float sum = (s[0][r] + s[1][r]) + (s[2][r] + s[3][r]);
            sum = redsum16(sum);
            lrun[r] = lrun[r] * alpha[r] + sum;
        }
#pragma unroll
        for (int tn = 0; tn < 4; ++tn)
#pragma unroll
            for (int r = 0; r < 8; ++r)
                o[tn][r] *= alpha[r];

        // P: C-layout -> A-layout via wave-private LDS slice
#pragma unroll
        for (int tj = 0; tj < 4; ++tj)
#pragma unroll
            for (int r = 0; r < 8; ++r)
                Ps[wave][half * 8 + r][tj * 16 + l16] = (__bf16)s[tj][r];

        Frag pa[2];
#pragma unroll
        for (int c = 0; c < 2; ++c) {
            const __bf16* p = &Ps[wave][l16][c * 32 + half * 16];
            pa[c].q[0] = *(const uint4*)p;
            pa[c].q[1] = *(const uint4*)(p + 8);
        }

#pragma unroll
        for (int tn = 0; tn < 4; ++tn) {
#pragma unroll
            for (int c = 0; c < 2; ++c) {
                Frag bv;
                const __bf16* p = Vt + headV + (size_t)(tn * 16 + l16) * TSEQ
                                  + kBase + c * 32 + half * 16;
                bv.q[0] = *(const uint4*)p;
                bv.q[1] = *(const uint4*)(p + 8);
                o[tn] = wmma_bf16(pa[c].v, bv.v, o[tn]);
            }
        }
    }

#pragma unroll
    for (int r = 0; r < 8; ++r) {
        float inv = 1.0f / lrun[r];
        int q = qBase + half * 8 + r;
#pragma unroll
        for (int tn = 0; tn < 4; ++tn) {
            int d = tn * 16 + l16;
            Oattn[(size_t)(b * TSEQ + q) * CDIM + h * HDD + d] = (__bf16)(o[tn][r] * inv);
        }
    }
}

// ---------------------------------------------------------------------------
extern "C" void kernel_launch(void* const* d_in, const int* in_sizes, int n_in,
                              void* d_out, int out_size, void* d_ws, size_t ws_size,
                              hipStream_t stream) {
    (void)in_sizes; (void)n_in; (void)out_size; (void)ws_size;
    const float* x  = (const float*)d_in[0];
    const float* Wq = (const float*)d_in[1];
    const float* Wk = (const float*)d_in[2];
    const float* Wv = (const float*)d_in[3];
    const float* Wo = (const float*)d_in[4];
    float* out = (float*)d_out;
    char* ws = (char*)d_ws;

    const size_t szX   = (size_t)BT * CDIM * sizeof(__bf16);    // 16 MB
    const size_t szW   = (size_t)CDIM * CDIM * sizeof(__bf16);  //  2 MB
    const size_t szQKV = (size_t)NB * NH * TSEQ * HDD * sizeof(__bf16); // 16 MB

    __bf16* xb  = (__bf16*)(ws);
    __bf16* wqb = (__bf16*)(ws + szX);
    __bf16* wkb = (__bf16*)(ws + szX + szW);
    __bf16* wvb = (__bf16*)(ws + szX + 2 * szW);
    __bf16* wob = (__bf16*)(ws + szX + 3 * szW);
    __bf16* Qb  = (__bf16*)(ws + szX + 4 * szW);
    __bf16* Kb  = (__bf16*)(ws + szX + 4 * szW + szQKV);
    __bf16* Vtb = (__bf16*)(ws + szX + 4 * szW + 2 * szQKV);
    __bf16* Atb = (__bf16*)(ws + szX + 4 * szW + 3 * szQKV);    // [B,T,C] bf16

    dim3 blk(256);
    // one-time f32 -> bf16
    int n4x = BT * CDIM / 4, n4w = CDIM * CDIM / 4;
    cvt_f32_bf16<<<(n4x + 255) / 256, blk, 0, stream>>>(x,  xb,  n4x);
    cvt_f32_bf16<<<(n4w + 255) / 256, blk, 0, stream>>>(Wq, wqb, n4w);
    cvt_f32_bf16<<<(n4w + 255) / 256, blk, 0, stream>>>(Wk, wkb, n4w);
    cvt_f32_bf16<<<(n4w + 255) / 256, blk, 0, stream>>>(Wv, wvb, n4w);
    cvt_f32_bf16<<<(n4w + 255) / 256, blk, 0, stream>>>(Wo, wob, n4w);

    dim3 gG(BT / 128, CDIM / 128);   // 64 x 8
    gemm_nt_bf16wmma<<<gG, blk, 0, stream>>>(xb, wqb, Qb, 0.125f, 0, BT); // 1/sqrt(HD)
    gemm_nt_bf16wmma<<<gG, blk, 0, stream>>>(xb, wkb, Kb, 1.0f, 1, BT);
    gemm_nt_bf16wmma<<<gG, blk, 0, stream>>>(xb, wvb, Vtb, 1.0f, 2, BT);

    dim3 gA(TSEQ / 128, NB * NH);    // 16 x 64
    flash_attn_bf16<<<gA, blk, 0, stream>>>(Qb, Kb, Vtb, Atb);

    gemm_nt_bf16wmma<<<gG, blk, 0, stream>>>(Atb, wob, out, 1.0f, 3, BT);
}
